// EnhancedLocalAttentionWithGQA_64536178589855
// MI455X (gfx1250) — matrix operand order repaired and checked by reading, not compiled, not run on hardware
//
#include <hip/hip_runtime.h>

// ---------------------------------------------------------------------------
// Types / helpers
// ---------------------------------------------------------------------------
typedef __attribute__((ext_vector_type(16))) __bf16 v16bf;
typedef __attribute__((ext_vector_type(8)))  float  v8f;
typedef __attribute__((ext_vector_type(4)))  int    v4i;
typedef unsigned short u16;

union FragB { v16bf v; uint4 u[2]; };

#if __has_builtin(__builtin_amdgcn_global_load_async_to_lds_b128)
#define USE_ASYNC_LDS 1
#else
#define USE_ASYNC_LDS 0
#endif

__device__ __forceinline__ u16 f2bf(float f) {
  unsigned int u = __float_as_uint(f);
  u += 0x7FFFu + ((u >> 16) & 1u);          // round-to-nearest-even
  return (u16)(u >> 16);
}

__device__ __forceinline__ v8f zero8() {
  v8f z = {0.f, 0.f, 0.f, 0.f, 0.f, 0.f, 0.f, 0.f};
  return z;
}

__device__ __forceinline__ v8f wmma_bf16f32(v16bf a, v16bf b, v8f c) {
  // (neg_a, A, neg_b, B, c_mod, C, reuse_a, reuse_b)
  return __builtin_amdgcn_wmma_f32_16x16x32_bf16(false, a, false, b, (short)0, c,
                                                 false, false);
}

__device__ __forceinline__ void wait_async0() {
#if USE_ASYNC_LDS
#if __has_builtin(__builtin_amdgcn_s_wait_asynccnt)
  __builtin_amdgcn_s_wait_asynccnt(0);
#else
  asm volatile("s_wait_asynccnt 0x0" ::: "memory");
#endif
#endif
}

// Copy one 16B chunk global -> LDS (async DMA when available).
__device__ __forceinline__ void copy16_g2l(u16* l, const u16* g) {
#if USE_ASYNC_LDS
  __builtin_amdgcn_global_load_async_to_lds_b128(
      (__attribute__((address_space(1))) v4i*)(v4i*)(void*)g,
      (__attribute__((address_space(3))) v4i*)(v4i*)(void*)l, 0, 0);
#else
  *reinterpret_cast<uint4*>(l) = *reinterpret_cast<const uint4*>(g);
#endif
}

// ---------------------------------------------------------------------------
// Tiled bf16 GEMM:  C(MxN, f32) = A(MxK bf16, row-major) * BT(NxK bf16)^T
// Block: 256 threads (8 waves, 2x4), tile 128x128, k-step 32, double-buffered
// LDS with async global->LDS staging.
// ---------------------------------------------------------------------------
__device__ __forceinline__ void stage_tiles(const u16* __restrict__ A,
                                            const u16* __restrict__ BT,
                                            u16* as, u16* bs,
                                            int mBase, int nBase, int K, int kb,
                                            int tid) {
#if !USE_ASYNC_LDS
  uint4 va[2], vb[2];
#pragma unroll
  for (int i = 0; i < 2; ++i) {             // batch loads first (overlap latency)
    const int c = tid + i * 256;
    const int row = c >> 2;
    const int kc = (c & 3) * 8;
    va[i] = *reinterpret_cast<const uint4*>(A + (size_t)(mBase + row) * K + kb + kc);
    vb[i] = *reinterpret_cast<const uint4*>(BT + (size_t)(nBase + row) * K + kb + kc);
  }
#pragma unroll
  for (int i = 0; i < 2; ++i) {
    const int c = tid + i * 256;
    const int row = c >> 2;
    const int kc = (c & 3) * 8;
    *reinterpret_cast<uint4*>(as + row * 32 + kc) = va[i];
    *reinterpret_cast<uint4*>(bs + row * 32 + kc) = vb[i];
  }
#else
#pragma unroll
  for (int i = 0; i < 2; ++i) {
    const int c = tid + i * 256;            // 512 chunks of 8 bf16 per tile
    const int row = c >> 2;
    const int kc = (c & 3) * 8;
    copy16_g2l(as + row * 32 + kc, A + (size_t)(mBase + row) * K + kb + kc);
    copy16_g2l(bs + row * 32 + kc, BT + (size_t)(nBase + row) * K + kb + kc);
  }
#endif
}

__global__ __launch_bounds__(256) void gemm_bf16_kernel(
    const u16* __restrict__ A, const u16* __restrict__ BT,
    const float* __restrict__ bias, float* __restrict__ C,
    int M, int N, int K, int addBias)
{
  __shared__ u16 As[2][128 * 32];
  __shared__ u16 Bs[2][128 * 32];
  const int tid  = threadIdx.x;
  const int lane = tid & 31;
  const int wid  = tid >> 5;
  const int lh   = lane >> 4;      // lane half (0/1)
  const int l16  = lane & 15;
  const int mBase = blockIdx.y * 128;
  const int nBase = blockIdx.x * 128;
  const int mw0 = (wid >> 2) * 64; // 4 M-tiles per wave
  const int nw0 = (wid & 3) * 32;  // 2 N-tiles per wave

  v8f acc[4][2];
#pragma unroll
  for (int mi = 0; mi < 4; ++mi)
#pragma unroll
    for (int nj = 0; nj < 2; ++nj) acc[mi][nj] = zero8();

  stage_tiles(A, BT, As[0], Bs[0], mBase, nBase, K, 0, tid);
  wait_async0();
  __syncthreads();

  int buf = 0;
  for (int kb = 0; kb < K; kb += 32) {
    const bool hasNext = (kb + 32) < K;
    if (hasNext)
      stage_tiles(A, BT, As[buf ^ 1], Bs[buf ^ 1], mBase, nBase, K, kb + 32, tid);

    FragB Af[4];
#pragma unroll
    for (int mi = 0; mi < 4; ++mi) {
      const uint4* p =
          reinterpret_cast<const uint4*>(&As[buf][(mw0 + mi * 16 + l16) * 32]);
      Af[mi].u[0] = p[lh];        // K = lh*8 .. lh*8+7
      Af[mi].u[1] = p[lh + 2];    // K = lh*8+16 .. lh*8+23
    }
    FragB Bf[2];
#pragma unroll
    for (int nj = 0; nj < 2; ++nj) {
      const uint4* p = reinterpret_cast<const uint4*>(
          &Bs[buf][(nw0 + nj * 16 + l16) * 32]);
      Bf[nj].u[0] = p[lh];        // K = lh*8 .. lh*8+7
      Bf[nj].u[1] = p[lh + 2];    // K = lh*8+16 .. lh*8+23
    }
#pragma unroll
    for (int mi = 0; mi < 4; ++mi)
#pragma unroll
      for (int nj = 0; nj < 2; ++nj)
        acc[mi][nj] = wmma_bf16f32(Af[mi].v, Bf[nj].v, acc[mi][nj]);

    if (hasNext) wait_async0();   // my async writes to buf^1 are done
    __syncthreads();              // everyone done reading buf / writing buf^1
    buf ^= 1;
  }

#pragma unroll
  for (int mi = 0; mi < 4; ++mi)
#pragma unroll
    for (int nj = 0; nj < 2; ++nj)
#pragma unroll
      for (int r = 0; r < 8; ++r) {
        const int m  = mBase + mw0 + mi * 16 + lh * 8 + r;
        const int nn = nBase + nw0 + nj * 16 + l16;
        float v = acc[mi][nj][r];
        if (addBias) v += bias[nn];
        C[(size_t)m * N + nn] = v;
      }
}

// ---------------------------------------------------------------------------
// Flash-style windowed attention, one block per (window n, head h, batch b).
// Q: [B,H,L,D] bf16, K/V: [B,G,L,D] bf16, O: [B, T=L, H, D] f32.
// Window n: queries/keys at positions n*128 .. n*128+255; output t = n*256+w.
// ---------------------------------------------------------------------------
__global__ __launch_bounds__(256) void attn_flash_kernel(
    const u16* __restrict__ Q, const u16* __restrict__ Kk,
    const u16* __restrict__ V, float* __restrict__ O)
{
  const int n = blockIdx.x, h = blockIdx.y, b = blockIdx.z;
  const int g    = h & 1;          // GQA: G = 2
  const int lane = threadIdx.x & 31;
  const int wid  = threadIdx.x >> 5;
  const int lh   = lane >> 4;
  const int l16  = lane & 15;

  __shared__ u16 Vt[64 * 256];     // V transposed: [d][k]   (32 KB)
  __shared__ u16 Pl[8 * 32 * 32];  // per-wave P staging      (16 KB)

  const int pos0 = n * 128;
  const u16* qb = Q  + (((size_t)b * 16 + h) * 4096 + pos0) * 64;
  const u16* kb = Kk + (((size_t)b * 2  + g) * 4096 + pos0) * 64;
  const u16* vb = V  + (((size_t)b * 2  + g) * 4096 + pos0) * 64;

  { // stage V^T: thread t handles key k = t
    const int k = threadIdx.x;
    const uint4* vr = reinterpret_cast<const uint4*>(vb + (size_t)k * 64);
#pragma unroll
    for (int c = 0; c < 8; ++c) {
      uint4 uq = vr[c];
      unsigned int wv[4] = {uq.x, uq.y, uq.z, uq.w};
#pragma unroll
      for (int jj = 0; jj < 4; ++jj) {
        Vt[(c * 8 + jj * 2 + 0) * 256 + k] = (u16)(wv[jj] & 0xffffu);
        Vt[(c * 8 + jj * 2 + 1) * 256 + k] = (u16)(wv[jj] >> 16);
      }
    }
  }
  __syncthreads();

  // Q fragments for this wave's 32 query rows (2 M-tiles x 2 k-chunks of D=64)
  FragB Qf[2][2];
  const int q0 = wid * 32;
#pragma unroll
  for (int mi = 0; mi < 2; ++mi) {
    const uint4* qr =
        reinterpret_cast<const uint4*>(qb + (size_t)(q0 + mi * 16 + l16) * 64);
#pragma unroll
    for (int kc = 0; kc < 2; ++kc) {
      Qf[mi][kc].u[0] = qr[kc * 4 + lh];
      Qf[mi][kc].u[1] = qr[kc * 4 + lh + 2];
    }
  }

  v8f   Oa[2][4];
  float rmax[2][8], rsum[2][8];
#pragma unroll
  for (int mi = 0; mi < 2; ++mi) {
#pragma unroll
    for (int nt = 0; nt < 4; ++nt) Oa[mi][nt] = zero8();
#pragma unroll
    for (int r = 0; r < 8; ++r) { rmax[mi][r] = -1e30f; rsum[mi][r] = 0.f; }
  }

  for (int c = 0; c < 8; ++c) {    // 8 key chunks of 32
    FragB Kf[2][2];                // B-frags for S: K rows are already N x K
#pragma unroll
    for (int nj = 0; nj < 2; ++nj) {
      const uint4* krp = reinterpret_cast<const uint4*>(
          kb + (size_t)(c * 32 + nj * 16 + l16) * 64);
#pragma unroll
      for (int kc = 0; kc < 2; ++kc) {
        Kf[nj][kc].u[0] = krp[kc * 4 + lh];
        Kf[nj][kc].u[1] = krp[kc * 4 + lh + 2];
      }
    }
    v8f S[2][2];
#pragma unroll
    for (int mi = 0; mi < 2; ++mi)
#pragma unroll
      for (int nj = 0; nj < 2; ++nj) {
        v8f a = zero8();
        a = wmma_bf16f32(Qf[mi][0].v, Kf[nj][0].v, a);
        a = wmma_bf16f32(Qf[mi][1].v, Kf[nj][1].v, a);
#pragma unroll
        for (int r = 0; r < 8; ++r) a[r] *= 0.125f;   // 1/sqrt(64)
        S[mi][nj] = a;
      }

#pragma unroll
    for (int mi = 0; mi < 2; ++mi) {
      float mc[8], ps[8], fs[8];
#pragma unroll
      for (int r = 0; r < 8; ++r) mc[r] = fmaxf(S[mi][0][r], S[mi][1][r]);
#pragma unroll
      for (int sh = 8; sh >= 1; sh >>= 1)
#pragma unroll
        for (int r = 0; r < 8; ++r)
          mc[r] = fmaxf(mc[r], __shfl_xor(mc[r], sh, 16));
#pragma unroll
      for (int r = 0; r < 8; ++r) {
        float mn   = fmaxf(rmax[mi][r], mc[r]);
        fs[r]      = __expf(rmax[mi][r] - mn);
        rmax[mi][r] = mn;
        float p0 = __expf(S[mi][0][r] - mn);
        float p1 = __expf(S[mi][1][r] - mn);
        S[mi][0][r] = p0; S[mi][1][r] = p1;
        ps[r] = p0 + p1;
      }
#pragma unroll
      for (int sh = 8; sh >= 1; sh >>= 1)
#pragma unroll
        for (int r = 0; r < 8; ++r) ps[r] += __shfl_xor(ps[r], sh, 16);
#pragma unroll
      for (int r = 0; r < 8; ++r) rsum[mi][r] = rsum[mi][r] * fs[r] + ps[r];
#pragma unroll
      for (int nt = 0; nt < 4; ++nt)
#pragma unroll
        for (int r = 0; r < 8; ++r) Oa[mi][nt][r] *= fs[r];

      u16* pw = Pl + wid * 1024;   // restage P (C-layout -> row-major 32x32)
#pragma unroll
      for (int nj = 0; nj < 2; ++nj)
#pragma unroll
        for (int r = 0; r < 8; ++r)
          pw[(mi * 16 + lh * 8 + r) * 32 + nj * 16 + l16] = f2bf(S[mi][nj][r]);
    }
    __syncthreads();

    FragB Pa[2];                   // A-frags of P (M=32 queries, K=32 keys)
#pragma unroll
    for (int mi = 0; mi < 2; ++mi) {
      const uint4* pr =
          reinterpret_cast<const uint4*>(Pl + wid * 1024 + (mi * 16 + l16) * 32);
      Pa[mi].u[0] = pr[lh];
      Pa[mi].u[1] = pr[lh + 2];
    }
    FragB Vf[4];                   // B-frags of V from Vt (N=d, K=keys)
#pragma unroll
    for (int nt = 0; nt < 4; ++nt) {
      const uint4* vtp = reinterpret_cast<const uint4*>(
          Vt + (size_t)(nt * 16 + l16) * 256 + c * 32);
      Vf[nt].u[0] = vtp[lh];
      Vf[nt].u[1] = vtp[lh + 2];
    }
#pragma unroll
    for (int mi = 0; mi < 2; ++mi)
#pragma unroll
      for (int nt = 0; nt < 4; ++nt)
        Oa[mi][nt] = wmma_bf16f32(Pa[mi].v, Vf[nt].v, Oa[mi][nt]);
    __syncthreads();
  }

  float* ob = O + (((size_t)b * 4096 + (size_t)n * 256) * 16 + h) * 64;
#pragma unroll
  for (int mi = 0; mi < 2; ++mi)
#pragma unroll
    for (int r = 0; r < 8; ++r) {
      const float inv = 1.0f / rsum[mi][r];
      const int   t   = q0 + mi * 16 + lh * 8 + r;
#pragma unroll
      for (int nt = 0; nt < 4; ++nt)
        ob[(size_t)t * 1024 + nt * 16 + l16] = Oa[mi][nt][r] * inv;
    }
}

// ---------------------------------------------------------------------------
// Elementwise / prep kernels
// ---------------------------------------------------------------------------
__global__ void f32_to_bf16_kernel(const float* __restrict__ in,
                                   u16* __restrict__ out, int n) {
  int i = blockIdx.x * blockDim.x + threadIdx.x;
  if (i < n) out[i] = f2bf(in[i]);
}

// W (K x N f32, row-major) -> WT (N x K bf16, row-major)
__global__ void transpose_bf16_kernel(const float* __restrict__ in,
                                      u16* __restrict__ out, int K, int N) {
  int k = blockIdx.x * 16 + threadIdx.x;
  int n = blockIdx.y * 16 + threadIdx.y;
  if (k < K && n < N) out[(size_t)n * K + k] = f2bf(in[(size_t)k * N + n]);
}

// RoPE: in (B*L, nh*64) f32 -> out [B, nh, L, 64] bf16 (concat-halves variant)
__global__ void rope_kernel(const float* __restrict__ in,
                            u16* __restrict__ out, int nh) {
  int idx = blockIdx.x * blockDim.x + threadIdx.x;  // B*L*nh*32 threads
  int j   = idx & 31;
  int tmp = idx >> 5;
  int h   = tmp % nh;
  int row = tmp / nh;        // 0..8191
  int pos = row & 4095;
  int b   = row >> 12;
  const float* src = in + (size_t)row * ((size_t)nh * 64) + (size_t)h * 64;
  float qe = src[2 * j];
  float qo = src[2 * j + 1];
  float dv = __expf(-0.2878231366242558f * (float)j);  // exp(-ln(1e4)*2j/64)
  float s, cth;
  sincosf((float)pos * dv, &s, &cth);
  size_t ob = (((size_t)b * nh + h) * 4096 + pos) * 64;
  out[ob + j]      = f2bf(qe * cth - qo * s);
  out[ob + 32 + j] = f2bf(qe * s + qo * cth);
}

// v (B*L, G*64) f32 -> [B, G, L, 64] bf16
__global__ void reorder_v_kernel(const float* __restrict__ in,
                                 u16* __restrict__ out) {
  int idx = blockIdx.x * blockDim.x + threadIdx.x;  // B*L*128
  int d   = idx & 63;
  int g   = (idx >> 6) & 1;
  int row = idx >> 7;
  int pos = row & 4095;
  int b   = row >> 12;
  out[(((size_t)b * 2 + g) * 4096 + pos) * 64 + d] = f2bf(in[idx]);
}

__global__ void lambda_kernel(const float* lq1, const float* lk1,
                              const float* lq2, const float* lk2, float* lam) {
  if (threadIdx.x == 0) {
    float d1 = 0.f, d2 = 0.f;
    for (int i = 0; i < 64; ++i) { d1 += lq1[i] * lk1[i]; d2 += lq2[i] * lk2[i]; }
    float v = __expf(d1) - __expf(d2) + 0.8f;
    *lam = fminf(0.9f, fmaxf(0.1f, v));
  }
}

__global__ void combine_kernel(const float* __restrict__ O1,
                               const float* __restrict__ O2,
                               const float* __restrict__ lam,
                               u16* __restrict__ out, int n) {
  int i = blockIdx.x * blockDim.x + threadIdx.x;
  if (i < n) out[i] = f2bf(O1[i] - (*lam) * O2[i]);
}

// ---------------------------------------------------------------------------
// Host orchestration
// ---------------------------------------------------------------------------
extern "C" void kernel_launch(void* const* d_in, const int* in_sizes, int n_in,
                              void* d_out, int out_size, void* d_ws, size_t ws_size,
                              hipStream_t stream) {
  (void)in_sizes; (void)n_in; (void)out_size; (void)ws_size;
  const float* x    = (const float*)d_in[0];
  const float* Wq1  = (const float*)d_in[1];
  const float* Wq2  = (const float*)d_in[2];
  const float* Wk1  = (const float*)d_in[3];
  const float* Wk2  = (const float*)d_in[4];
  const float* Wv   = (const float*)d_in[5];
  const float* Wout = (const float*)d_in[6];
  const float* bout = (const float*)d_in[7];
  const float* lq1  = (const float*)d_in[8];
  const float* lk1  = (const float*)d_in[9];
  const float* lq2  = (const float*)d_in[10];
  const float* lk2  = (const float*)d_in[11];

  const int M = 8192;        // B*L
  const int E = 1024, KV = 128;

  char*  base = (char*)d_ws;
  size_t off  = 0;
  auto take = [&](size_t bytes) -> char* {
    char* p = base + off;
    off += (bytes + 255) & ~(size_t)255;
    return p;
  };

  float* lam     = (float*)take(4);
  u16*   xb      = (u16*)take((size_t)M * E * 2);       // also reused as attnC
  u16*   Wq1T    = (u16*)take((size_t)E * E * 2);
  u16*   Wq2T    = (u16*)take((size_t)E * E * 2);
  u16*   Wk1T    = (u16*)take((size_t)KV * E * 2);
  u16*   Wk2T    = (u16*)take((size_t)KV * E * 2);
  u16*   WvT     = (u16*)take((size_t)KV * E * 2);
  u16*   WoutT   = (u16*)take((size_t)E * E * 2);
  float* tmpBig  = (float*)take((size_t)M * E * 4);     // q f32 temp; reused as O1
  float* tmpSml  = (float*)take((size_t)M * KV * 4);    // k/v f32 temp
  u16*   q1r     = (u16*)take((size_t)M * E * 2);       // [B,H,L,D]
  u16*   q2r     = (u16*)take((size_t)M * E * 2);
  u16*   k1r     = (u16*)take((size_t)M * KV * 2);      // [B,G,L,D]
  u16*   k2r     = (u16*)take((size_t)M * KV * 2);
  u16*   vbf     = (u16*)take((size_t)M * KV * 2);
  float* O2      = (float*)take((size_t)M * E * 4);
  float* O1      = tmpBig;
  u16*   attnC   = xb;

  // 1) lambda scalar
  lambda_kernel<<<1, 32, 0, stream>>>(lq1, lk1, lq2, lk2, lam);

  // 2) convert x to bf16
  f32_to_bf16_kernel<<<(M * E) / 256, 256, 0, stream>>>(x, xb, M * E);

  // 3) transpose+convert weights to N x K bf16
  transpose_bf16_kernel<<<dim3(64, 64), dim3(16, 16), 0, stream>>>(Wq1, Wq1T, E, E);
  transpose_bf16_kernel<<<dim3(64, 64), dim3(16, 16), 0, stream>>>(Wq2, Wq2T, E, E);
  transpose_bf16_kernel<<<dim3(64, 8),  dim3(16, 16), 0, stream>>>(Wk1, Wk1T, E, KV);
  transpose_bf16_kernel<<<dim3(64, 8),  dim3(16, 16), 0, stream>>>(Wk2, Wk2T, E, KV);
  transpose_bf16_kernel<<<dim3(64, 8),  dim3(16, 16), 0, stream>>>(Wv,  WvT,  E, KV);
  transpose_bf16_kernel<<<dim3(64, 64), dim3(16, 16), 0, stream>>>(Wout, WoutT, E, E);

  // 4) projections + RoPE / reorder
  gemm_bf16_kernel<<<dim3(E / 128, M / 128), 256, 0, stream>>>(
      xb, Wq1T, (const float*)nullptr, tmpBig, M, E, E, 0);
  rope_kernel<<<(M * 16 * 32) / 256, 256, 0, stream>>>(tmpBig, q1r, 16);

  gemm_bf16_kernel<<<dim3(E / 128, M / 128), 256, 0, stream>>>(
      xb, Wq2T, (const float*)nullptr, tmpBig, M, E, E, 0);
  rope_kernel<<<(M * 16 * 32) / 256, 256, 0, stream>>>(tmpBig, q2r, 16);

  gemm_bf16_kernel<<<dim3(KV / 128, M / 128), 256, 0, stream>>>(
      xb, Wk1T, (const float*)nullptr, tmpSml, M, KV, E, 0);
  rope_kernel<<<(M * 2 * 32) / 256, 256, 0, stream>>>(tmpSml, k1r, 2);

  gemm_bf16_kernel<<<dim3(KV / 128, M / 128), 256, 0, stream>>>(
      xb, Wk2T, (const float*)nullptr, tmpSml, M, KV, E, 0);
  rope_kernel<<<(M * 2 * 32) / 256, 256, 0, stream>>>(tmpSml, k2r, 2);

  gemm_bf16_kernel<<<dim3(KV / 128, M / 128), 256, 0, stream>>>(
      xb, WvT, (const float*)nullptr, tmpSml, M, KV, E, 0);
  reorder_v_kernel<<<(M * KV) / 256, 256, 0, stream>>>(tmpSml, vbf);

  // 5) windowed attention: only windows 0..15 reach the output (t = n*256 + w)
  attn_flash_kernel<<<dim3(16, 16, 2), 256, 0, stream>>>(q1r, k1r, vbf, O1);
  attn_flash_kernel<<<dim3(16, 16, 2), 256, 0, stream>>>(q2r, k2r, vbf, O2);

  // 6) a = a1 - lambda * a2  (bf16, laid out as (B*L, E) with e = h*64+d)
  combine_kernel<<<(M * E) / 256, 256, 0, stream>>>(O1, O2, lam, attnC, M * E);

  // 7) out = a @ Wout + bout  (f32 output)
  gemm_bf16_kernel<<<dim3(E / 128, M / 128), 256, 0, stream>>>(
      attnC, WoutT, bout, (float*)d_out, M, E, E, 1);
}
